// GroupBertAttention_19052474925331
// MI455X (gfx1250) — compile-verified, matrix operand order checked by reading, not run on hardware
//
#include <hip/hip_runtime.h>
#include <hip/hip_bf16.h>

// ---------------------------------------------------------------------------
// GroupBertAttention for MI455X (gfx1250): bf16 WMMA everywhere, fused
// flash-attention (no S x S scores buffer), fp32 softmax/LN/residual.
// GEMMs use the Tensor Data Mover (tensor_load_to_lds) with double-buffered
// LDS tiles and TDM pad-insertion for bank-conflict-free ds_load_b128 reads.
// ---------------------------------------------------------------------------

typedef unsigned short u16;
typedef __attribute__((ext_vector_type(16))) __bf16 v16bf;
typedef __attribute__((ext_vector_type(8)))  float  v8f;
typedef __attribute__((ext_vector_type(4)))  unsigned uv4;
typedef __attribute__((ext_vector_type(8)))  int      iv8;
typedef __attribute__((ext_vector_type(4)))  int      iv4;

static constexpr int Bz = 2;      // batch
static constexpr int S  = 2048;   // sequence
static constexpr int D  = 1024;   // model dim
static constexpr int H  = 16;     // heads
static constexpr int HD = 64;     // head dim
static constexpr int M  = Bz * S; // 4096 rows
static constexpr float LOG2E = 1.44269504088896340736f;

// LDS tile geometry: K-step = 32 bf16 = 64B per row; TDM pads +16B per row.
static constexpr int ROWB   = 80;              // padded LDS row stride (bytes)
static constexpr int ATILEB = 64 * ROWB;       // 5120 B
static constexpr int BTILEB = 256 * ROWB;      // 20480 B

union FragBF {
    uint4 q[2];
    u16   s[16];
    v16bf v;
};

__device__ __forceinline__ u16 f2bf(float x) {
    union { float f; unsigned u; } v; v.f = x;
    unsigned r = v.u + 0x7fffu + ((v.u >> 16) & 1u);   // round-to-nearest-even
    return (u16)(r >> 16);
}

// ---------------------------------------------------------------------------
// Global 16x32 bf16 WMMA fragment load (A- or B-side; mirror layouts).
// lane<16 -> row=lane, K chunks {0..7, 16..23}; lane>=16 -> row=lane-16,
// K chunks {8..15, 24..31}.  Two contiguous 16B global_load_b128 per lane.
// ---------------------------------------------------------------------------
__device__ __forceinline__ v16bf load_frag16(const u16* __restrict__ base,
                                             int ld, int row0, int k0) {
    const int lane = threadIdx.x & 31;
    const int l  = lane & 15;
    const int hl = lane >> 4;
    const u16* p = base + (size_t)(row0 + l) * ld + k0 + hl * 8;
    FragBF f;
    f.q[0] = *(const uint4*)(p);
    f.q[1] = *(const uint4*)(p + 16);
    return f.v;
}

// Same fragment pattern out of a padded LDS tile (row stride ROWB bytes).
__device__ __forceinline__ v16bf lds_frag16(const char* base, int row0) {
    const int lane = threadIdx.x & 31;
    const int l  = lane & 15;
    const int hl = lane >> 4;
    const char* p = base + (row0 + l) * ROWB + hl * 16;
    FragBF f;
    f.q[0] = *(const uint4*)(p);        // K elems {0..7} or {8..15}
    f.q[1] = *(const uint4*)(p + 32);   // K elems {16..23} or {24..31}
    return f.v;
}

__device__ __forceinline__ v8f wmma_bf16(v16bf a, v16bf b, v8f c) {
    return __builtin_amdgcn_wmma_f32_16x16x32_bf16(
        /*neg_a=*/false, a, /*neg_b=*/false, b,
        /*c_mod=*/(short)0, c, /*reuse_a=*/false, /*reuse_b=*/false);
}

// ---------------------------------------------------------------------------
// Tensor Data Mover: 2D bf16 tile (tile_d0 x tile_d1 elements) from global
// row-major (leading dim stride0 elements) into LDS at lds_off, inserting
// 16B of padding after every 64B row (pad_interval=16 DW, pad_amount=4 DW)
// so LDS rows are ROWB bytes apart.
// ---------------------------------------------------------------------------
__device__ __forceinline__ void tdm_load_2d(unsigned lds_off, const void* gptr,
                                            unsigned tile_d0, unsigned tile_d1,
                                            unsigned stride0) {
    unsigned long long ga = (unsigned long long)(size_t)gptr;
    uv4 g0;
    g0.x = 1u;                                           // count=1 (valid), user
    g0.y = lds_off;                                      // LDS byte address
    g0.z = (unsigned)ga;                                 // global_addr[31:0]
    g0.w = (unsigned)((ga >> 32) & 0x1FFFFFFu) | (2u << 30); // [56:32] + type=2
    iv8 g1;
    g1[0] = (int)((1u << 16) |        // data_size = 1 -> 2 bytes
                  (1u << 20) |        // pad_enable
                  (3u << 22) |        // pad_interval: 16 DWORDs (64B)
                  (3u << 25));        // pad_amount:   4 DWORDs (16B)
    g1[1] = (int)((tile_d0 & 0xFFFFu) << 16);            // tensor_dim0[15:0]
    g1[2] = (int)((tile_d0 >> 16) | ((tile_d1 & 0xFFFFu) << 16)); // dim0 hi|dim1 lo
    g1[3] = (int)((tile_d1 >> 16) | (tile_d0 << 16));    // dim1 hi | tile_dim0
    g1[4] = (int)(tile_d1 & 0xFFFFu);                    // tile_dim1 (tile_dim2=0)
    g1[5] = (int)stride0;                                // tensor_dim0_stride lo
    g1[6] = 0;                                           // stride hi / dim1 stride
    g1[7] = 0;
    iv4 z4 = {0, 0, 0, 0};
#if __clang_major__ >= 23
    iv8 z8 = {0, 0, 0, 0, 0, 0, 0, 0};
    __builtin_amdgcn_tensor_load_to_lds(g0, g1, z4, z4, z8, 0);
#else
    __builtin_amdgcn_tensor_load_to_lds(g0, g1, z4, z4, 0);
#endif
}

// ---------------------------------------------------------------------------
// Kernel 1: weights fp32 -> bf16.  Wqkv = concat(wq,wk,wv) rows, wo separate.
// ---------------------------------------------------------------------------
__global__ void wconv_kernel(const float* __restrict__ wq, const float* __restrict__ wk,
                             const float* __restrict__ wv, const float* __restrict__ wo,
                             u16* __restrict__ wqkv, u16* __restrict__ wob) {
    size_t i = (size_t)blockIdx.x * 256 + threadIdx.x;   // 4,194,304 total
    if (i < (size_t)3 * D * D) {
        int n = (int)(i >> 10);
        float v = (n < D) ? wq[i] : (n < 2 * D) ? wk[i - (size_t)D * D]
                                                : wv[i - (size_t)2 * D * D];
        wqkv[i] = f2bf(v);
    } else {
        size_t j = i - (size_t)3 * D * D;
        wob[j] = f2bf(wo[j]);
    }
}

// ---------------------------------------------------------------------------
// Kernel 2: LayerNorm (one block per row) -> bf16 activations.
// ---------------------------------------------------------------------------
__global__ __launch_bounds__(256) void ln_kernel(const float* __restrict__ x,
                                                 const float* __restrict__ g,
                                                 const float* __restrict__ be,
                                                 u16* __restrict__ xbf) {
    const int row = blockIdx.x;
    const int t = threadIdx.x, lane = t & 31, wid = t >> 5;
    const float* xr = x + (size_t)row * D;
    float4 v = *(const float4*)(xr + t * 4);
    float s  = v.x + v.y + v.z + v.w;
    float ss = v.x * v.x + v.y * v.y + v.z * v.z + v.w * v.w;
#pragma unroll
    for (int off = 16; off >= 1; off >>= 1) {
        s  += __shfl_xor(s,  off, 32);
        ss += __shfl_xor(ss, off, 32);
    }
    __shared__ float rs[8], rss[8];
    if (lane == 0) { rs[wid] = s; rss[wid] = ss; }
    __syncthreads();
    float Sm = 0.f, SSm = 0.f;
#pragma unroll
    for (int w = 0; w < 8; ++w) { Sm += rs[w]; SSm += rss[w]; }
    const float mu   = Sm * (1.0f / D);
    const float var  = SSm * (1.0f / D) - mu * mu;
    const float rstd = rsqrtf(var + 1e-12f);
    u16* o = xbf + (size_t)row * D + t * 4;
    const float xs[4] = { v.x, v.y, v.z, v.w };
#pragma unroll
    for (int c = 0; c < 4; ++c) {
        float y = (xs[c] - mu) * rstd * g[t * 4 + c] + be[t * 4 + c];
        o[c] = f2bf(y);
    }
}

// ---------------------------------------------------------------------------
// Kernels 3 & 5: NT-GEMM  Y = A(bf16,[M,K]) * W(bf16,[N,K])^T  via WMMA with
// double-buffered TDM tiles in LDS.
// Block = 8 waves (2 along M x 4 along N); block tile 64x256; K-step 32.
// Wave 0 DMAs the 64x32 A tile, wave 1 DMAs the 256x32 W tile (TENSORcnt).
// MODE 0: QKV  -> scatter epilogue + bias to q[b,h,s,d], k[b,h,s,d], vT[b,h,d,s]
// MODE 1: OUT  -> fp32 epilogue: out = acc + bo + residual(hidden)
// ---------------------------------------------------------------------------
template <int MODE>
__global__ __launch_bounds__(256) void gemm_kernel(
    const u16* __restrict__ A, const u16* __restrict__ W, int K,
    const float* __restrict__ bq, const float* __restrict__ bk, const float* __restrict__ bv,
    u16* __restrict__ qbuf, u16* __restrict__ kbuf, u16* __restrict__ vtbuf,
    const float* __restrict__ bo, const float* __restrict__ hid, float* __restrict__ out) {
    __shared__ __align__(128) char smA[2][ATILEB];
    __shared__ __align__(128) char smB[2][BTILEB];

    const int lane = threadIdx.x & 31, wid = threadIdx.x >> 5;
    const int wm = wid & 1, wn = wid >> 1;
    const int mb = blockIdx.y * 64;
    const int nb = blockIdx.x * 256;

    v8f acc[2][4];
#pragma unroll
    for (int i = 0; i < 2; ++i)
#pragma unroll
        for (int j = 0; j < 4; ++j)
            acc[i][j] = (v8f){0.f, 0.f, 0.f, 0.f, 0.f, 0.f, 0.f, 0.f};

    const int nk = K / 32;
    // Prologue: DMA first K-tile into buffer 0.
    if (wid == 0)
        tdm_load_2d((unsigned)(size_t)(void*)&smA[0][0], A + (size_t)mb * K, 32, 64, K);
    else if (wid == 1)
        tdm_load_2d((unsigned)(size_t)(void*)&smB[0][0], W + (size_t)nb * K, 32, 256, K);

    for (int it = 0; it < nk; ++it) {
        const int cur = it & 1;
        // All waves done reading the back buffer before TDM overwrites it.
        asm volatile("s_wait_dscnt 0x0" ::: "memory");
        __syncthreads();
        if (it + 1 < nk) {
            const int nxt = cur ^ 1;
            const int k0 = (it + 1) * 32;
            if (wid == 0)
                tdm_load_2d((unsigned)(size_t)(void*)&smA[nxt][0],
                            A + (size_t)mb * K + k0, 32, 64, K);
            else if (wid == 1)
                tdm_load_2d((unsigned)(size_t)(void*)&smB[nxt][0],
                            W + (size_t)nb * K + k0, 32, 256, K);
            if (wid < 2) __builtin_amdgcn_s_wait_tensorcnt(1); // tile `it` done
        } else {
            if (wid < 2) __builtin_amdgcn_s_wait_tensorcnt(0);
        }
        __syncthreads();    // buf[cur] is fully in LDS for every wave

        v16bf af[2], bfr[4];
#pragma unroll
        for (int i = 0; i < 2; ++i) af[i]  = lds_frag16(&smA[cur][0], wm * 32 + 16 * i);
#pragma unroll
        for (int j = 0; j < 4; ++j) bfr[j] = lds_frag16(&smB[cur][0], wn * 64 + 16 * j);
#pragma unroll
        for (int i = 0; i < 2; ++i)
#pragma unroll
            for (int j = 0; j < 4; ++j)
                acc[i][j] = wmma_bf16(af[i], bfr[j], acc[i][j]);
    }

    // Epilogue: C layout -> row = v + 8*(lane>=16), col = lane&15.
    const int l = lane & 15, hl = lane >> 4;
    const int m0 = mb + wm * 32;
    const int n0 = nb + wn * 64;
#pragma unroll
    for (int i = 0; i < 2; ++i) {
#pragma unroll
        for (int j = 0; j < 4; ++j) {
#pragma unroll
            for (int v = 0; v < 8; ++v) {
                const int gm = m0 + i * 16 + hl * 8 + v;
                const int gn = n0 + j * 16 + l;
                float val = acc[i][j][v];
                if (MODE == 0) {
                    const int b = gm >> 11, s = gm & (S - 1);
                    const int seg = gn >> 10, c = gn & (D - 1);
                    const int h = c >> 6, d = c & (HD - 1);
                    val += (seg == 0) ? bq[c] : (seg == 1) ? bk[c] : bv[c];
                    const u16 bv16 = f2bf(val);
                    const size_t bh = (size_t)(b * H + h);
                    if (seg == 0)      qbuf[(bh * S + s) * HD + d]   = bv16;
                    else if (seg == 1) kbuf[(bh * S + s) * HD + d]   = bv16;
                    else               vtbuf[(bh * HD + d) * S + s]  = bv16;
                } else {
                    out[(size_t)gm * D + gn] = val + bo[gn] + hid[(size_t)gm * D + gn];
                }
            }
        }
    }
}

// ---------------------------------------------------------------------------
// Kernel 4: fused flash attention.  Per wave: 16 query rows, loop 32 keys/iter.
// S^T = K_tile * Q^T  (rows = keys -> softmax reductions mostly per-lane),
// ctx^T = V^T * P^T    (P^T C-layout repacks into a B-fragment per-lane).
// ---------------------------------------------------------------------------
__global__ __launch_bounds__(256) void attn_kernel(
    const u16* __restrict__ qbuf, const u16* __restrict__ kbuf,
    const u16* __restrict__ vtbuf, const float* __restrict__ mask,
    u16* __restrict__ ctx) {
    const int lane = threadIdx.x & 31, wid = threadIdx.x >> 5;
    const int b = blockIdx.z, h = blockIdx.y;
    const int q0 = blockIdx.x * 128 + wid * 16;
    const size_t bh = (size_t)(b * H + h);
    const u16* qh = qbuf + bh * S * HD;
    const u16* kh = kbuf + bh * S * HD;
    const u16* vh = vtbuf + bh * HD * S;
    const float* mp = mask + (size_t)b * S;
    const int l = lane & 15, hl = lane >> 4;
    const float SC = 0.125f * LOG2E;   // 1/sqrt(64) folded into exp2 domain

    // Q^T B-fragments (lane = query column), loaded once.
    const v16bf bq0 = load_frag16(qh, HD, q0, 0);
    const v16bf bq1 = load_frag16(qh, HD, q0, 32);

    v8f acco[4];
#pragma unroll
    for (int t = 0; t < 4; ++t) acco[t] = (v8f){0.f,0.f,0.f,0.f,0.f,0.f,0.f,0.f};
    float mrun = -INFINITY, lrun = 0.f;
    const v8f vz = (v8f){0.f,0.f,0.f,0.f,0.f,0.f,0.f,0.f};

    for (int kb = 0; kb < S; kb += 32) {
        // ---- scores^T: two 16-key tiles, K-dim = head dim (64) = 2 steps ----
        v8f s0 = wmma_bf16(load_frag16(kh, HD, kb,      0),  bq0, vz);
        s0     = wmma_bf16(load_frag16(kh, HD, kb,      32), bq1, s0);
        v8f s1 = wmma_bf16(load_frag16(kh, HD, kb + 16, 0),  bq0, vz);
        s1     = wmma_bf16(load_frag16(kh, HD, kb + 16, 32), bq1, s1);

        float sc[16];
#pragma unroll
        for (int v = 0; v < 8; ++v) {
            sc[v]     = s0[v] * SC + mp[kb +      hl * 8 + v] * LOG2E;
            sc[8 + v] = s1[v] * SC + mp[kb + 16 + hl * 8 + v] * LOG2E;
        }
        // ---- online softmax (per query column = per lane, + one shuffle) ----
        float mx = sc[0];
#pragma unroll
        for (int e = 1; e < 16; ++e) mx = fmaxf(mx, sc[e]);
        mx = fmaxf(mx, __shfl_xor(mx, 16, 32));
        const float mnew  = fmaxf(mrun, mx);
        const float alpha = exp2f(mrun - mnew);
        float rs = 0.f;
#pragma unroll
        for (int e = 0; e < 16; ++e) { float p = exp2f(sc[e] - mnew); sc[e] = p; rs += p; }
        rs += __shfl_xor(rs, 16, 32);
        lrun = lrun * alpha + rs;
        mrun = mnew;

        // ---- P^T (C-layout) -> bf16 B-fragment: pure per-lane repack ----
        FragBF pf;
#pragma unroll
        for (int e = 0; e < 8; ++e) { pf.s[e] = f2bf(sc[e]); pf.s[8 + e] = f2bf(sc[8 + e]); }

#pragma unroll
        for (int t = 0; t < 4; ++t) acco[t] = acco[t] * alpha;
#pragma unroll
        for (int t = 0; t < 4; ++t) {
            const v16bf av = load_frag16(vh, S, t * 16, kb);  // V^T: rows=dims
            acco[t] = wmma_bf16(av, pf.v, acco[t]);
        }
    }

    const float inv = 1.0f / lrun;
    const int q = q0 + l;
#pragma unroll
    for (int t = 0; t < 4; ++t)
#pragma unroll
        for (int v = 0; v < 8; ++v) {
            const int d = t * 16 + hl * 8 + v;
            ctx[((size_t)b * S + q) * D + h * HD + d] = f2bf(acco[t][v] * inv);
        }
}

// ---------------------------------------------------------------------------
// Host launcher
// ---------------------------------------------------------------------------
extern "C" void kernel_launch(void* const* d_in, const int* in_sizes, int n_in,
                              void* d_out, int out_size, void* d_ws, size_t ws_size,
                              hipStream_t stream) {
    const float* hidden = (const float*)d_in[0];
    const float* amask  = (const float*)d_in[1];
    const float* gamma  = (const float*)d_in[2];
    const float* beta   = (const float*)d_in[3];
    const float* wq = (const float*)d_in[4];  const float* bq = (const float*)d_in[5];
    const float* wk = (const float*)d_in[6];  const float* bk = (const float*)d_in[7];
    const float* wv = (const float*)d_in[8];  const float* bv = (const float*)d_in[9];
    const float* wo = (const float*)d_in[10]; const float* bo = (const float*)d_in[11];
    float* out = (float*)d_out;

    // Workspace carve-up (bytes), all 256B-aligned chunks.
    char* ws = (char*)d_ws;
    u16* xbf   = (u16*)ws;                       ws += (size_t)M * D * 2;        // 8 MB
    u16* wqkvb = (u16*)ws;                       ws += (size_t)3 * D * D * 2;    // 6 MB
    u16* wob   = (u16*)ws;                       ws += (size_t)D * D * 2;        // 2 MB
    u16* qb    = (u16*)ws;                       ws += (size_t)M * D * 2;        // 8 MB
    u16* kbuf  = (u16*)ws;                       ws += (size_t)M * D * 2;        // 8 MB
    u16* vtb   = (u16*)ws;                       ws += (size_t)M * D * 2;        // 8 MB
    u16* ctx   = (u16*)ws;                       ws += (size_t)M * D * 2;        // 8 MB

    // 1) weights -> bf16
    wconv_kernel<<<(4 * D * D) / 256, 256, 0, stream>>>(wq, wk, wv, wo, wqkvb, wob);
    // 2) LayerNorm -> bf16 activations
    ln_kernel<<<M, 256, 0, stream>>>(hidden, gamma, beta, xbf);
    // 3) fused QKV GEMM (N = 3072) with scatter epilogue
    gemm_kernel<0><<<dim3(3 * D / 256, M / 64), 256, 0, stream>>>(
        xbf, wqkvb, D, bq, bk, bv, qb, kbuf, vtb, nullptr, nullptr, nullptr);
    // 4) fused flash attention
    attn_kernel<<<dim3(S / 128, H, Bz), 256, 0, stream>>>(qb, kbuf, vtb, amask, ctx);
    // 5) output projection + bias + residual (fp32 out)
    gemm_kernel<1><<<dim3(D / 256, M / 64), 256, 0, stream>>>(
        ctx, wob, D, nullptr, nullptr, nullptr, nullptr, nullptr, nullptr,
        bo, hidden, out);
}
